// SubbandFilter_64922725646643
// MI455X (gfx1250) — matrix operand order repaired and checked by reading, not compiled, not run on hardware
//
#include <hip/hip_runtime.h>

// ---------------------------------------------------------------------------
// Subband filter bank on MI455X (gfx1250), formulated as a bf16 WMMA GEMM.
//
//   y[chan, band, n] = sum_t w_band[t] * xpad[chan, 8n + t]        (K = 48000)
//
// xpad = x with 23999 leading zeros (centered FIR). w has 16 columns:
// 8 bands x {re, im}; the im column is a windowed-Hilbert FIR of the band
// filter (the Hilbert operator is LTI and commutes with convolution).
//
// A (16x32 bf16)  : Toeplitz slice of xpad, A[m,k] = xpad[8*(n0+m)+kBase+k]
// B (32x16 bf16)  : B[k,n] = wT[n][kBase+k], staged through LDS with CDNA5
//                   async global->LDS copies (double-buffered), shared by all
//                   4 row tiles of the wave.
// D (16x16 f32)   : 16 decimated samples x 16 band-part columns.
//
// Everything (x: 5.7MB bf16, filters: 1.5MB bf16, out: 21MB) is L2-resident
// (192MB); the kernel is tensor-pipe/L0 bound, so the design minimizes
// memory instructions per WMMA: 8 global A + 2 LDS B loads per 4 WMMAs.
// ---------------------------------------------------------------------------

typedef __attribute__((ext_vector_type(16))) __bf16 v16bf;
typedef __attribute__((ext_vector_type(8)))  __bf16 v8bf;
typedef __attribute__((ext_vector_type(8)))  float  v8f;

#define SBF_L      661500
#define SBF_N      47999
#define SBF_KPAD   48000
#define SBF_LEAD   23999
#define SBF_RLEN   709504            // >= 8*(Ld-1) + KPAD, multiple of 8
#define SBF_LD     82688             // ceil(661500 / 8) ; 82688 = 323*256
#define SBF_NCH    4                 // B*C
#define SBF_PI     3.14159265358979323846f

#define SBF_WT_BYTES   (16u * SBF_KPAD * 2u)              // 1,536,000 (256-aligned)
// workspace total ~7.2 MB (wT + xpad)

#define SBF_WAVES  4
#define SBF_MT     4                 // row tiles per wave (share one B operand)
#define SBF_KS     640               // K-slab staged in LDS (75 slabs * 640 = 48000)
#define SBF_KSP    648               // padded column stride: 324 dwords = 4 mod 64
#define SBF_NSLAB  75

// Async global->LDS staging path: the clang builtin uses HIP language-AS
// pointer params we cannot spell portably, so use the ISA instruction via
// inline asm (VDST = per-lane LDS byte offset, VADDR = per-lane 64b address).
#define SBF_ASYNC 1

typedef __attribute__((address_space(3))) void as3_void;

static __device__ __forceinline__ void sbf_wait_async() {
#if SBF_ASYNC
    asm volatile("s_wait_asynccnt 0x0" ::: "memory");
#endif
}

static __device__ __forceinline__
void sbf_async_copy_b128(const __bf16* g, __bf16* l) {
#if SBF_ASYNC
    unsigned loff = (unsigned)(unsigned long long)(as3_void*)l;  // LDS byte offset
    asm volatile("global_load_async_to_lds_b128 %0, %1, off"
                 :: "v"(loff), "v"(g) : "memory");
#else
    *(v8bf*)l = *(const v8bf*)g;          // global_load_b128 + ds_store_b128
#endif
}

// --------------------------- filter construction ---------------------------
// wT[col][t], col-major, stride KPAD. col = band (re) or band+8 (im).
// Taps stored time-reversed so the GEMM K index walks xpad forward.
__global__ __launch_bounds__(256)
void sbf_build_filters(const float* __restrict__ v, __bf16* __restrict__ wT) {
    int id = blockIdx.x * blockDim.x + threadIdx.x;
    if (id >= 8 * SBF_KPAD) return;
    int band = id / SBF_KPAD;
    int t    = id % SBF_KPAD;
    float re = 0.f, im = 0.f;
    if (t < SBF_N) {
        int m = SBF_N - 1 - t;               // time-reverse
        const float* vb = v + band * SBF_N;
        re = vb[m];
        // Windowed Hilbert FIR: h[j] = 2/(pi*j) for odd j, Hamming window.
        #pragma unroll 4
        for (int j = 1; j <= 127; j += 2) {
            float w  = 0.54f + 0.46f * __cosf(SBF_PI * (float)j / 128.0f);
            float cj = (2.0f / (SBF_PI * (float)j)) * w;
            float xm = (m - j >= 0)    ? vb[m - j] : 0.f;
            float xp = (m + j < SBF_N) ? vb[m + j] : 0.f;
            im += cj * (xm - xp);
        }
    }
    wT[(size_t)band * SBF_KPAD + t]       = (__bf16)re;
    wT[(size_t)(band + 8) * SBF_KPAD + t] = (__bf16)im;
}

// ------------------------------ signal padding -----------------------------
__global__ __launch_bounds__(256)
void sbf_pad_signal(const float* __restrict__ x, __bf16* __restrict__ xpad) {
    int id = blockIdx.x * blockDim.x + threadIdx.x;
    if (id >= SBF_NCH * SBF_RLEN) return;
    int ch  = id / SBF_RLEN;
    int p   = id % SBF_RLEN;
    int src = p - SBF_LEAD;
    float val = (src >= 0 && src < SBF_L) ? x[(size_t)ch * SBF_L + src] : 0.f;
    xpad[id] = (__bf16)val;
}

// ------------------------- B-slab staging into LDS -------------------------
// 16 cols x 640 bf16 = 1280 b128 chunks; 128 threads -> 10 chunks/thread
// (each async instruction moves 32 lanes x 16B = 512B per wave).
static __device__ __forceinline__
void sbf_stage_slab(__bf16* lbuf, const __bf16* __restrict__ wT,
                    int kbase, int tid) {
    #pragma unroll
    for (int j = 0; j < 10; ++j) {
        int c   = tid + j * 128;
        int col = c / 80;                 // 80 chunks per column
        int ko  = (c - col * 80) * 8;
        const __bf16* g = wT + (size_t)col * SBF_KPAD + kbase + ko;
        __bf16*       l = lbuf + col * SBF_KSP + ko;
        sbf_async_copy_b128(g, l);
    }
}

// ------------------------------- main GEMM ---------------------------------
// 128 threads = 4 waves. Each wave owns MT=4 row tiles (64 samples), so a
// workgroup covers 256 decimated samples x 16 band-part columns (exact:
// 82688 = 323 * 256 -> no boundary guards in the hot loop).
__global__ __launch_bounds__(128)
void sbf_subband_wmma(const __bf16* __restrict__ xpad,
                      const __bf16* __restrict__ wT,
                      const float*  __restrict__ fcenter,
                      const int*    __restrict__ srp,
                      const int*    __restrict__ facp,
                      float* __restrict__ out) {
    __shared__ __align__(16) __bf16 ldsB[2][16 * SBF_KSP];   // 41,472 B

    const int tid  = threadIdx.x;
    const int lane = tid & 31;
    const int wave = tid >> 5;
    const int chan = blockIdx.y;
    const int n0   = blockIdx.x * (SBF_WAVES * SBF_MT * 16) + wave * (SBF_MT * 16);

    const int sel = lane >> 4;            // 0: lanes 0-15, 1: lanes 16-31
    const int m   = lane & 15;            // A row within tile / B column

    const __bf16* xrow = xpad + (size_t)chan * SBF_RLEN;
    // A lane base per tile: element 8*(n0 + mt*16 + m) + sel*8 ; chunks +k,+k+16
    const __bf16* aB[SBF_MT];
    #pragma unroll
    for (int mt = 0; mt < SBF_MT; ++mt)
        aB[mt] = xrow + 8 * (n0 + mt * 16 + m) + sel * 8;

    // B lane base in LDS: column m, K chunk sel*8 ; chunks +kc, +kc+16
    const __bf16* bcol0 = &ldsB[0][m * SBF_KSP + sel * 8];
    const __bf16* bcol1 = &ldsB[1][m * SBF_KSP + sel * 8];

    v8f acc[SBF_MT];
    #pragma unroll
    for (int mt = 0; mt < SBF_MT; ++mt) acc[mt] = (v8f){};

    sbf_stage_slab(&ldsB[0][0], wT, 0, tid);
    sbf_wait_async();
    __syncthreads();

    for (int s = 0; s < SBF_NSLAB; ++s) {
        if (s + 1 < SBF_NSLAB)
            sbf_stage_slab(&ldsB[(s + 1) & 1][0], wT, (s + 1) * SBF_KS, tid);

        const __bf16* bcol = (s & 1) ? bcol1 : bcol0;
        const int kg = s * SBF_KS;

        for (int kc = 0; kc < SBF_KS; kc += 32) {
            v8bf bLo = *(const v8bf*)(bcol + kc);        // ds_load_b128
            v8bf bHi = *(const v8bf*)(bcol + kc + 16);   // ds_load_b128
            v16bf B  = __builtin_shufflevector(bLo, bHi,
                         0,1,2,3,4,5,6,7,8,9,10,11,12,13,14,15);
            #pragma unroll
            for (int mt = 0; mt < SBF_MT; ++mt) {
                v8bf al = *(const v8bf*)(aB[mt] + kg + kc);       // global b128
                v8bf ah = *(const v8bf*)(aB[mt] + kg + kc + 16);  // global b128
                v16bf A = __builtin_shufflevector(al, ah,
                            0,1,2,3,4,5,6,7,8,9,10,11,12,13,14,15);
                acc[mt] = __builtin_amdgcn_wmma_f32_16x16x32_bf16(
                              false, A, false, B, (short)0, acc[mt],
                              false, false);
            }
        }
        // buffer (s+1)&1 must be complete, and everyone must be done reading
        // buffer s&1 before iteration s+1 stages into it.
        sbf_wait_async();
        __syncthreads();
    }

    // ---- epilogue: recombine re/im (lanes l and l^8), modulate, store ----
    const int band = lane & 7;
    const int part = (lane >> 3) & 1;     // 0 = re column, 1 = im column
    const float srf   = (float)srp[0];
    const float fac   = (float)facp[0];
    const float omega = -fcenter[band] * SBF_PI / (0.5f * srf);

    #pragma unroll
    for (int mt = 0; mt < SBF_MT; ++mt) {
        #pragma unroll
        for (int r = 0; r < 8; ++r) {
            int n = n0 + mt * 16 + r + 8 * sel;
            float mine  = acc[mt][r];
            float other = __shfl_xor(mine, 8, 32);
            float ph = omega * fac * (float)n;
            float s, c;
            __sincosf(ph, &s, &c);
            // part0 lane holds re, partner im:  re' = re*c - im*s
            // part1 lane holds im, partner re:  im' = re*s + im*c
            float val = (part == 0) ? (mine * c - other * s)
                                    : (other * s + mine * c);
            size_t idx = (((size_t)(chan * 8 + band)) * SBF_LD + n) * 2 + part;
            out[idx] = val;
        }
    }
}

// ------------------------------- launcher ----------------------------------
extern "C" void kernel_launch(void* const* d_in, const int* in_sizes, int n_in,
                              void* d_out, int out_size, void* d_ws, size_t ws_size,
                              hipStream_t stream) {
    const float* x       = (const float*)d_in[0];   // (2,2,661500) f32
    const float* v       = (const float*)d_in[1];   // (8,47999)    f32
    const float* fcenter = (const float*)d_in[2];   // (8,)         f32
    const int*   sr      = (const int*)d_in[3];     // scalar 44100
    const int*   fac     = (const int*)d_in[4];     // scalar 8

    __bf16* wT   = (__bf16*)d_ws;
    __bf16* xpad = (__bf16*)((char*)d_ws + SBF_WT_BYTES);

    {
        int n = 8 * SBF_KPAD;
        sbf_build_filters<<<(n + 255) / 256, 256, 0, stream>>>(v, wT);
    }
    {
        int n = SBF_NCH * SBF_RLEN;
        sbf_pad_signal<<<(n + 255) / 256, 256, 0, stream>>>(x, xpad);
    }
    {
        dim3 grid(SBF_LD / (SBF_WAVES * SBF_MT * 16), SBF_NCH);   // 323 x 4
        sbf_subband_wmma<<<grid, 128, 0, stream>>>(xpad, wT, fcenter, sr, fac,
                                                   (float*)d_out);
    }
}
